// CMoE_53274774339937
// MI455X (gfx1250) — matrix operand order repaired and checked by reading, not compiled, not run on hardware
//
#include <hip/hip_runtime.h>
#include <hip/hip_bf16.h>

// ---------------------------------------------------------------------------
// Multi-cell KAARMA recurrence on gfx1250 (MI455X).
//   B=512, T=2048, C=8 cells, M=100 dict entries, N=16 state dims.
// One workgroup = 16 batch rows, 8 waves (wave c == cell c).
// Both per-step contractions run as v_wmma_f32_16x16x32_bf16 with the
// Gaussian-kernel exponent fully folded into spare K rows, so
//   phi = exp2(2*log2e * acc).
// S / U / A live permanently in per-wave WMMA fragments (VGPRs).
// bf16 packing uses native (__bf16) casts -> v_cvt_pk_bf16_f32.
// ---------------------------------------------------------------------------

#define B_    512
#define T_    2048
#define C_    8
#define M_    100
#define N_    16
#define MT    7          // ceil(M/16) m-tiles (112 padded)
#define CHUNK 32         // x / gate staging chunk (T % CHUNK == 0)
#define LOG2E 1.44269504f

typedef __attribute__((ext_vector_type(16))) __bf16 v16bf;
typedef __attribute__((ext_vector_type(2)))  __bf16 v2bf;
typedef __attribute__((ext_vector_type(8)))  float  v8f;

union Frag {
    v16bf          v;
    __bf16         h[16];
    unsigned int   d[8];
    uint4          q[2];
};

union Pack2 {
    v2bf         v;
    unsigned int u;
};

__device__ __forceinline__ unsigned int pack_bf2(float a, float b) {
    Pack2 p;
    p.v[0] = (__bf16)a;    // native f32->bf16 convert (v_cvt_pk_bf16_f32)
    p.v[1] = (__bf16)b;
    return p.u;
}

__global__ __launch_bounds__(256)
void kaarma_scan_kernel(const float* __restrict__ x,   // [B, T]
                        const float* __restrict__ S,   // [C, M, N]
                        const float* __restrict__ U,   // [C, M]
                        const float* __restrict__ A,   // [C, M, N]
                        const float* __restrict__ W1,  // [1, 16]
                        const float* __restrict__ b1,  // [16]
                        const float* __restrict__ W2,  // [16, 8]
                        const float* __restrict__ b2,  // [8]
                        float* __restrict__ out)       // [B, T]
{
    // ---- LDS ---------------------------------------------------------------
    __shared__ float s_accum[16 * 16];                          // state accum [b][n]
    __shared__ float s_snorm[16];                               // ||state_b||^2
    __shared__ __align__(16) __bf16 s_stateBF[16 * 24];         // bf16 state, 48B row stride
    __shared__ __align__(16) __bf16 s_phi[C_ * 16 * 40];        // phi^T staging, 80B row stride
    __shared__ float s_x[16 * CHUNK];                           // x chunk [b][tt]
    __shared__ float s_gate[16 * CHUNK * C_];                   // gates [b][tt][c]
    __shared__ float s_mlp[16 + 16 + 128 + 8];                  // W1 | b1 | W2 | b2

    const int tid   = threadIdx.x;
    const int lane  = tid & 31;
    const int wid   = tid >> 5;          // wave id == cell id
    const int c     = wid;
    const int half  = lane >> 4;         // 0: lanes 0-15, 1: lanes 16-31
    const int lr    = lane & 15;
    const int bbase = blockIdx.x * 16;

    // ---- setup: zero LDS, stage MLP weights --------------------------------
    for (int i = tid; i < 16 * 16; i += 256) s_accum[i] = 0.f;
    if (tid < 16) s_snorm[tid] = 0.f;
    for (int i = tid; i < 16 * 24; i += 256) s_stateBF[i] = (__bf16)0.f;
    for (int i = tid; i < C_ * 16 * 40; i += 256) s_phi[i] = (__bf16)0.f;
    if (tid < 16)                 s_mlp[tid]       = W1[tid];
    else if (tid < 32)            s_mlp[tid]       = b1[tid - 16];
    else if (tid < 160)           s_mlp[tid]       = W2[tid - 32];
    else if (tid < 168)           s_mlp[tid]       = b2[tid - 160];

    // ---- persistent per-wave fragments -------------------------------------
    // G1 A-operand: S-chunk [16 m x 32 K], K16=U_m, K17=1, K18=cc_m
    Frag sfrag[MT];
#pragma unroll
    for (int t = 0; t < MT; ++t) {
        Frag f;
        int m = t * 16 + lr;
        bool valid = (m < M_);
        if (!half) {
#pragma unroll
            for (int k = 0; k < 8; ++k)
                f.h[k] = (__bf16)(valid ? S[(c * M_ + m) * N_ + k] : 0.f);
            float uv = valid ? U[c * M_ + m] : 0.f;
            float sn = 0.f;
            if (valid) {
#pragma unroll
                for (int k = 0; k < N_; ++k) {
                    float sv = S[(c * M_ + m) * N_ + k];
                    sn += sv * sv;
                }
            }
            float cc = valid ? (-0.5f * (sn + uv * uv)) : -30000.f;
            f.h[8]  = (__bf16)uv;      // K16 = U_m
            f.h[9]  = (__bf16)1.0f;    // K17 = 1
            f.h[10] = (__bf16)cc;      // K18 = -(||S_m||^2 + U_m^2)/2
            f.h[11] = f.h[12] = f.h[13] = f.h[14] = f.h[15] = (__bf16)0.f;
        } else {
#pragma unroll
            for (int k = 0; k < 8; ++k)
                f.h[k] = (__bf16)(valid ? S[(c * M_ + m) * N_ + 8 + k] : 0.f);
#pragma unroll
            for (int k = 8; k < 16; ++k) f.h[k] = (__bf16)0.f;  // K24..31 pad
        }
        sfrag[t] = f;
    }
    // G2 A-operand: A^T chunks [16 n x 32 m]
    Frag atfrag[4];
#pragma unroll
    for (int qk = 0; qk < 4; ++qk) {
        Frag f;
        int n = lr;
#pragma unroll
        for (int i = 0; i < 8; ++i) {
            int m0 = qk * 32 + (half ? 8 : 0) + i;        // K 0..7 / 8..15
            int m1 = qk * 32 + 16 + (half ? 8 : 0) + i;   // K 16..23 / 24..31
            f.h[i]     = (__bf16)((m0 < M_) ? A[(c * M_ + m0) * N_ + n] : 0.f);
            f.h[8 + i] = (__bf16)((m1 < M_) ? A[(c * M_ + m1) * N_ + n] : 0.f);
        }
        atfrag[qk] = f;
    }
    __syncthreads();

    // ---- main recurrence ----------------------------------------------------
    for (int t0 = 0; t0 < T_; t0 += CHUNK) {
        // stage x chunk + precompute gates (gate depends only on x)
        for (int i = tid; i < 16 * CHUNK; i += 256) {
            int b  = i / CHUNK;
            int tt = i % CHUNK;
            float xv = x[(size_t)(bbase + b) * T_ + t0 + tt];
            s_x[i] = xv;
            float lg[8];
#pragma unroll
            for (int cc2 = 0; cc2 < 8; ++cc2) lg[cc2] = s_mlp[160 + cc2];
#pragma unroll
            for (int j = 0; j < 16; ++j) {
                float h = fmaxf(xv * s_mlp[j] + s_mlp[16 + j], 0.f);
#pragma unroll
                for (int cc2 = 0; cc2 < 8; ++cc2)
                    lg[cc2] += h * s_mlp[32 + j * 8 + cc2];
            }
            float mx = lg[0];
#pragma unroll
            for (int cc2 = 1; cc2 < 8; ++cc2) mx = fmaxf(mx, lg[cc2]);
            float ev[8], ssum = 0.f;
#pragma unroll
            for (int cc2 = 0; cc2 < 8; ++cc2) {
                ev[cc2] = __expf(lg[cc2] - mx);
                ssum += ev[cc2];
            }
            float inv = 1.f / ssum;
#pragma unroll
            for (int cc2 = 0; cc2 < 8; ++cc2) s_gate[i * 8 + cc2] = ev[cc2] * inv;
        }
        __syncthreads();

        for (int ts = 0; ts < CHUNK; ++ts) {
            const int t = t0 + ts;

            // ---- build B operand for G1: state^T [32K x 16b] ---------------
            Frag bs;
            {
                const uint4* sp = (const uint4*)s_stateBF;   // 48B row stride = 3x uint4
                bs.q[0] = sp[lr * 3 + half];                 // n = 8*half..8*half+7
            }
            if (!half) {                                     // lanes 0-15 carry K16..23 folds
                float xb = s_x[lr * CHUNK + ts];
                float rr = -0.5f * (s_snorm[lr] + xb * xb);
                bs.d[4] = pack_bf2(xb, rr);                  // K16 = x_b, K17 = rowterm
                bs.d[5] = pack_bf2(1.0f, 0.f);               // K18 = 1
                bs.d[6] = 0u;
                bs.d[7] = 0u;
            } else {
#pragma unroll
                for (int i = 4; i < 8; ++i) bs.d[i] = 0u;
            }
            const float g = s_gate[(lr * CHUNK + ts) * 8 + c];

            // ---- G1 (7 WMMA) -> phi -> LDS -> G2 (4 WMMA), chunked ---------
            v8f ns = {};   // ns^T[n, b] accumulator
#pragma unroll
            for (int qk = 0; qk < 4; ++qk) {
                unsigned int* rowu =
                    (unsigned int*)(s_phi + (c * 16 + lr) * 40);
#pragma unroll
                for (int s = 0; s < 2; ++s) {
                    const int tt = 2 * qk + s;
                    if (tt < MT) {
                        v8f acc = {};
                        acc = __builtin_amdgcn_wmma_f32_16x16x32_bf16(
                            false, sfrag[tt].v, false, bs.v, (short)0, acc,
                            false, false);
#pragma unroll
                        for (int r = 0; r < 8; r += 2) {
                            float p0 = __builtin_amdgcn_exp2f(acc[r]     * (2.0f * LOG2E));
                            float p1 = __builtin_amdgcn_exp2f(acc[r + 1] * (2.0f * LOG2E));
                            rowu[s * 8 + (r >> 1) + half * 4] = pack_bf2(p0, p1);
                        }
                    } else {   // m-tile 7: zero pad cols 112..127
#pragma unroll
                        for (int r = 0; r < 8; r += 2)
                            rowu[s * 8 + (r >> 1) + half * 4] = 0u;
                    }
                }
                asm volatile("s_wait_dscnt 0x0" ::: "memory");
                Frag bp;   // B operand for G2: phi^T chunk [32 m x 16 b]
                {
                    const uint4* pp = (const uint4*)s_phi;   // 80B row = 5x uint4
                    bp.q[0] = pp[(c * 16 + lr) * 5 + half];
                    bp.q[1] = pp[(c * 16 + lr) * 5 + half + 2];
                }
                ns = __builtin_amdgcn_wmma_f32_16x16x32_bf16(
                    false, atfrag[qk].v, false, bp.v, (short)0, ns,
                    false, false);
            }

            // ---- gate-weighted reduction over cells via LDS f32 atomics ----
#pragma unroll
            for (int r = 0; r < 8; ++r) {
                int n = r + 8 * half;
                atomicAdd(&s_accum[lr * 16 + n], g * ns[r]);
            }
            __syncthreads();

            // ---- state rebuild (256 threads: tid -> (b, n)) ----------------
            {
                int b = tid >> 4, n = tid & 15;
                float v = s_accum[tid];
                s_accum[tid] = 0.f;                 // ready for next step
                float s2 = v * v;                   // ||state_b||^2 via shuffles
                s2 += __shfl_xor(s2, 1, 32);
                s2 += __shfl_xor(s2, 2, 32);
                s2 += __shfl_xor(s2, 4, 32);
                s2 += __shfl_xor(s2, 8, 32);
                if (n == 0) s_snorm[b] = s2;
                float vn = __shfl_xor(v, 1, 32);    // neighbour for bf16 pack
                if ((n & 1) == 0) {
                    *(unsigned int*)(s_stateBF + b * 24 + n) = pack_bf2(v, vn);
                }
                if (n == 15)
                    out[(size_t)(bbase + b) * T_ + t] = v;   // state[:, -1]
            }
            __syncthreads();
        }
    }
}

extern "C" void kernel_launch(void* const* d_in, const int* in_sizes, int n_in,
                              void* d_out, int out_size, void* d_ws, size_t ws_size,
                              hipStream_t stream) {
    (void)in_sizes; (void)n_in; (void)out_size; (void)d_ws; (void)ws_size;
    const float* x  = (const float*)d_in[0];
    const float* S  = (const float*)d_in[1];
    const float* U  = (const float*)d_in[2];
    const float* A  = (const float*)d_in[3];
    const float* W1 = (const float*)d_in[4];
    const float* b1 = (const float*)d_in[5];
    const float* W2 = (const float*)d_in[6];
    const float* b2 = (const float*)d_in[7];
    float* out = (float*)d_out;

    dim3 grid(B_ / 16);   // 32 workgroups of 16 batch rows
    dim3 block(256);      // 8 waves = 8 cells
    kaarma_scan_kernel<<<grid, block, 0, stream>>>(x, S, U, A, W1, b1, W2, b2, out);
}
